// MultiHeadAttention_19361712570393
// MI455X (gfx1250) — matrix operand order repaired and checked by reading, not compiled, not run on hardware
//
#include <hip/hip_runtime.h>
#include <math.h>

// Problem constants (match reference)
#define B_   4
#define S_   2048
#define D_   768
#define H_   12
#define DK_  64
#define NEG_ (-1e9f)
#define EPS_ (1e-5f)

typedef __attribute__((ext_vector_type(16))) __bf16 v16bf;
typedef __attribute__((ext_vector_type(8)))  float  v8f;

// ---------------------------------------------------------------------------
// Wave-level 32x32 bf16 WMMA step (2x2 tiles of v_wmma_f32_16x16x32_bf16).
// As : [rows][40] bf16 LDS tile, K-major (32 valid k per row)
// Bst: [cols][40] bf16 LDS tile, N-major (transposed; 32 valid k per col)
// VGPR layouts per CDNA5 ISA 7.12.2:
//   A (16x32 bf16): lane<16 holds row M=lane, k = {0..7,16..23}; lane>=16 same
//                   row, k = {8..15,24..31}  -> two contiguous 16B runs.
//   B (32x16 bf16): lane holds col N=lane%16, same k set as A:
//                   lane<16 k = {0..7,16..23}; lane>=16 k = {8..15,24..31}.
// ---------------------------------------------------------------------------
__device__ __forceinline__ void wave_mma_2x2(const __bf16 (*As)[40],
                                             const __bf16 (*Bst)[40],
                                             int m_off, int n_off, int lane,
                                             v8f acc[2][2]) {
  const int half = lane >> 4;
  const int l16  = lane & 15;
  union U { uint4 q[2]; v16bf v; };
  U a[2], b[2];
  const int kb  = half * 8;
  const int kb2 = half * 16;
#pragma unroll
  for (int i = 0; i < 2; ++i) {
    const __bf16* r = As[m_off + i * 16 + l16];
    a[i].q[0] = *(const uint4*)(r + kb);
    a[i].q[1] = *(const uint4*)(r + kb + 16);
  }
#pragma unroll
  for (int j = 0; j < 2; ++j) {
    const __bf16* r = Bst[n_off + j * 16 + l16];
    b[j].q[0] = *(const uint4*)(r + kb);
    b[j].q[1] = *(const uint4*)(r + kb + 16);
  }
#pragma unroll
  for (int i = 0; i < 2; ++i)
#pragma unroll
    for (int j = 0; j < 2; ++j)
      acc[i][j] = __builtin_amdgcn_wmma_f32_16x16x32_bf16(
          false, a[i].v, false, b[j].v, (short)0, acc[i][j], false, false);
}

// ---------------------------------------------------------------------------
// f32 -> bf16 conversion
// ---------------------------------------------------------------------------
__global__ void __launch_bounds__(256)
cvt_f32_to_bf16(const float* __restrict__ src, __bf16* __restrict__ dst,
                long n) {
  long i = (long)blockIdx.x * 256 + threadIdx.x;
  if (i < n) dst[i] = (__bf16)src[i];
}

// ---------------------------------------------------------------------------
// QKV projection: X[8192x768] @ W[768x768] + bias -> head-major bf16
// out[((b*H+h)*S+s)*64 + d]
// ---------------------------------------------------------------------------
__global__ void __launch_bounds__(256)
gemm_proj_head(const __bf16* __restrict__ X, const __bf16* __restrict__ W,
               const float* __restrict__ bias, __bf16* __restrict__ outHead) {
  __shared__ __attribute__((aligned(16))) __bf16 As[128][40];
  __shared__ __attribute__((aligned(16))) __bf16 Bst[64][40];
  const int tid = threadIdx.x, lane = tid & 31, wid = tid >> 5;
  const int m0 = blockIdx.y * 128, n0 = blockIdx.x * 64;
  const int m_off = (wid & 3) * 32, n_off = (wid >> 2) * 32;
  v8f z = {0.f, 0.f, 0.f, 0.f, 0.f, 0.f, 0.f, 0.f};
  v8f acc[2][2] = {{z, z}, {z, z}};

  for (int k0 = 0; k0 < D_; k0 += 32) {
    // A tile: 128 rows x 32 k, coalesced 16B chunks
#pragma unroll
    for (int t = tid; t < 512; t += 256) {
      const int row = t >> 2, kc = (t & 3) * 8;
      *(uint4*)&As[row][kc] =
          *(const uint4*)(X + (size_t)(m0 + row) * D_ + k0 + kc);
    }
    // B tile transposed: Bst[n][k] = W[k0+k][n0+n]
    {
      const int n = tid >> 2, kc = (tid & 3) * 8;
#pragma unroll
      for (int i = 0; i < 8; ++i)
        Bst[n][kc + i] = W[(size_t)(k0 + kc + i) * D_ + n0 + n];
    }
    if (k0 + 32 < D_)
      __builtin_prefetch(X + (size_t)(m0 + (tid >> 1)) * D_ + k0 + 32, 0, 1);
    __syncthreads();
    wave_mma_2x2(As, Bst, m_off, n_off, lane, acc);
    __syncthreads();
  }

  const int half = lane >> 4, l16 = lane & 15;
#pragma unroll
  for (int i = 0; i < 2; ++i)
#pragma unroll
    for (int j = 0; j < 2; ++j)
#pragma unroll
      for (int g = 0; g < 8; ++g) {
        const int m = m0 + m_off + i * 16 + half * 8 + g;
        const int n = n0 + n_off + j * 16 + l16;
        const float val = acc[i][j][g] + bias[n];
        const int bb = m >> 11, s = m & (S_ - 1);
        const int h = n >> 6, d = n & 63;
        outHead[(((size_t)(bb * H_ + h)) * S_ + s) * 64 + d] = (__bf16)val;
      }
}

// ---------------------------------------------------------------------------
// Attention scores: per (b,h) q[2048x64] @ k^T -> f32, scaled + pad-masked,
// written directly into the attn output region (pre-softmax).
// ---------------------------------------------------------------------------
__global__ void __launch_bounds__(256)
attn_scores(const __bf16* __restrict__ qh, const __bf16* __restrict__ kh,
            const unsigned char* __restrict__ pad, float* __restrict__ scores) {
  __shared__ __attribute__((aligned(16))) __bf16 As[128][40];
  __shared__ __attribute__((aligned(16))) __bf16 Bst[64][40];
  const int tid = threadIdx.x, lane = tid & 31, wid = tid >> 5;
  const int bh = blockIdx.z, b = bh / H_;
  const __bf16* q = qh + (size_t)bh * S_ * 64;
  const __bf16* kp = kh + (size_t)bh * S_ * 64;
  const int m0 = blockIdx.y * 128, n0 = blockIdx.x * 64;
  const int m_off = (wid & 3) * 32, n_off = (wid >> 2) * 32;
  v8f z = {0.f, 0.f, 0.f, 0.f, 0.f, 0.f, 0.f, 0.f};
  v8f acc[2][2] = {{z, z}, {z, z}};

#pragma unroll
  for (int k0 = 0; k0 < DK_; k0 += 32) {
#pragma unroll
    for (int t = tid; t < 512; t += 256) {
      const int row = t >> 2, kc = (t & 3) * 8;
      *(uint4*)&As[row][kc] =
          *(const uint4*)(q + (size_t)(m0 + row) * 64 + k0 + kc);
    }
    {
      // B = K^T, so Bst[n][k] = K[n0+n][k0+k] : contiguous 16B per thread
      const int n = tid >> 2, kc = (tid & 3) * 8;
      *(uint4*)&Bst[n][kc] =
          *(const uint4*)(kp + (size_t)(n0 + n) * 64 + k0 + kc);
    }
    __syncthreads();
    wave_mma_2x2(As, Bst, m_off, n_off, lane, acc);
    __syncthreads();
  }

  const int half = lane >> 4, l16 = lane & 15;
#pragma unroll
  for (int i = 0; i < 2; ++i)
#pragma unroll
    for (int j = 0; j < 2; ++j)
#pragma unroll
      for (int g = 0; g < 8; ++g) {
        const int m = m0 + m_off + i * 16 + half * 8 + g;
        const int n = n0 + n_off + j * 16 + l16;
        float val = acc[i][j][g] * 0.125f;  // 1/sqrt(64)
        if (pad[((size_t)b * S_ + m) * S_ + n]) val = NEG_;
        scores[((size_t)bh * S_ + m) * S_ + n] = val;
      }
}

// ---------------------------------------------------------------------------
// In-place row softmax over the attn tensor (rows of width 2048)
// ---------------------------------------------------------------------------
__global__ void __launch_bounds__(256)
softmax_rows(float* __restrict__ attn) {
  float* p = attn + (size_t)blockIdx.x * S_;
  const int tid = threadIdx.x;
  __shared__ float red[256];

  float lmax = -INFINITY;
  for (int i = tid; i < S_; i += 256) lmax = fmaxf(lmax, p[i]);
  red[tid] = lmax;
  __syncthreads();
  for (int s = 128; s > 0; s >>= 1) {
    if (tid < s) red[tid] = fmaxf(red[tid], red[tid + s]);
    __syncthreads();
  }
  const float m = red[0];
  __syncthreads();

  float lsum = 0.f;
  for (int i = tid; i < S_; i += 256) {
    const float e = __expf(p[i] - m);
    p[i] = e;
    lsum += e;
  }
  red[tid] = lsum;
  __syncthreads();
  for (int s = 128; s > 0; s >>= 1) {
    if (tid < s) red[tid] += red[tid + s];
    __syncthreads();
  }
  const float inv = 1.0f / red[0];
  for (int i = tid; i < S_; i += 256) p[i] *= inv;
}

// ---------------------------------------------------------------------------
// Context: per (b,h) attn[2048x2048](f32, converted to bf16 in LDS) @ v[2048x64]
// -> ctx bf16 in [B,S,H*64]
// ---------------------------------------------------------------------------
__global__ void __launch_bounds__(256)
attn_context(const float* __restrict__ attn, const __bf16* __restrict__ vh,
             __bf16* __restrict__ ctx) {
  __shared__ __attribute__((aligned(16))) __bf16 As[128][40];
  __shared__ __attribute__((aligned(16))) __bf16 Bst[64][40];
  const int tid = threadIdx.x, lane = tid & 31, wid = tid >> 5;
  const int bh = blockIdx.z, b = bh / H_, h = bh % H_;
  const float* a = attn + (size_t)bh * S_ * S_;
  const __bf16* v = vh + (size_t)bh * S_ * 64;
  const int m0 = blockIdx.y * 128;
  const int m_off = (wid & 3) * 32, n_off = (wid >> 2) * 32;  // N=64 exactly
  v8f z = {0.f, 0.f, 0.f, 0.f, 0.f, 0.f, 0.f, 0.f};
  v8f acc[2][2] = {{z, z}, {z, z}};

  for (int k0 = 0; k0 < S_; k0 += 32) {
    // A tile: f32 attn -> bf16 on the fly
#pragma unroll
    for (int t = tid; t < 512; t += 256) {
      const int row = t >> 2, kc = (t & 3) * 8;
      const float* src = a + (size_t)(m0 + row) * S_ + k0 + kc;
      const float4 f0 = *(const float4*)src;
      const float4 f1 = *(const float4*)(src + 4);
      __bf16* d = &As[row][kc];
      d[0] = (__bf16)f0.x; d[1] = (__bf16)f0.y;
      d[2] = (__bf16)f0.z; d[3] = (__bf16)f0.w;
      d[4] = (__bf16)f1.x; d[5] = (__bf16)f1.y;
      d[6] = (__bf16)f1.z; d[7] = (__bf16)f1.w;
    }
    // Bst[n][k] = v[k0+k][n]
    {
      const int n = tid >> 2, kc = (tid & 3) * 8;
#pragma unroll
      for (int i = 0; i < 8; ++i)
        Bst[n][kc + i] = v[(size_t)(k0 + kc + i) * 64 + n];
    }
    __syncthreads();
    wave_mma_2x2(As, Bst, m_off, n_off, lane, acc);
    __syncthreads();
  }

  const int half = lane >> 4, l16 = lane & 15;
#pragma unroll
  for (int i = 0; i < 2; ++i)
#pragma unroll
    for (int j = 0; j < 2; ++j)
#pragma unroll
      for (int g = 0; g < 8; ++g) {
        const int m = m0 + m_off + i * 16 + half * 8 + g;
        const int n = n_off + j * 16 + l16;
        ctx[((size_t)(b * S_ + m)) * D_ + h * 64 + n] = (__bf16)acc[i][j][g];
      }
}

// ---------------------------------------------------------------------------
// Output projection: ctx[8192x768] @ Wo[768x768] + bo + residual -> f32 tmp
// ---------------------------------------------------------------------------
__global__ void __launch_bounds__(256)
gemm_oproj(const __bf16* __restrict__ X, const __bf16* __restrict__ W,
           const float* __restrict__ bias, const float* __restrict__ resid,
           float* __restrict__ out) {
  __shared__ __attribute__((aligned(16))) __bf16 As[128][40];
  __shared__ __attribute__((aligned(16))) __bf16 Bst[64][40];
  const int tid = threadIdx.x, lane = tid & 31, wid = tid >> 5;
  const int m0 = blockIdx.y * 128, n0 = blockIdx.x * 64;
  const int m_off = (wid & 3) * 32, n_off = (wid >> 2) * 32;
  v8f z = {0.f, 0.f, 0.f, 0.f, 0.f, 0.f, 0.f, 0.f};
  v8f acc[2][2] = {{z, z}, {z, z}};

  for (int k0 = 0; k0 < D_; k0 += 32) {
#pragma unroll
    for (int t = tid; t < 512; t += 256) {
      const int row = t >> 2, kc = (t & 3) * 8;
      *(uint4*)&As[row][kc] =
          *(const uint4*)(X + (size_t)(m0 + row) * D_ + k0 + kc);
    }
    {
      const int n = tid >> 2, kc = (tid & 3) * 8;
#pragma unroll
      for (int i = 0; i < 8; ++i)
        Bst[n][kc + i] = W[(size_t)(k0 + kc + i) * D_ + n0 + n];
    }
    if (k0 + 32 < D_)
      __builtin_prefetch(X + (size_t)(m0 + (tid >> 1)) * D_ + k0 + 32, 0, 1);
    __syncthreads();
    wave_mma_2x2(As, Bst, m_off, n_off, lane, acc);
    __syncthreads();
  }

  const int half = lane >> 4, l16 = lane & 15;
#pragma unroll
  for (int i = 0; i < 2; ++i)
#pragma unroll
    for (int j = 0; j < 2; ++j)
#pragma unroll
      for (int g = 0; g < 8; ++g) {
        const int m = m0 + m_off + i * 16 + half * 8 + g;
        const int n = n0 + n_off + j * 16 + l16;
        out[(size_t)m * D_ + n] =
            acc[i][j][g] + bias[n] + resid[(size_t)m * D_ + n];
      }
}

// ---------------------------------------------------------------------------
// Row LayerNorm over width 768
// ---------------------------------------------------------------------------
__global__ void __launch_bounds__(256)
layernorm_rows(const float* __restrict__ x, const float* __restrict__ g,
               const float* __restrict__ bb, float* __restrict__ out) {
  const float* p = x + (size_t)blockIdx.x * D_;
  float* o = out + (size_t)blockIdx.x * D_;
  const int tid = threadIdx.x;
  __shared__ float red[256];

  float s = 0.f;
  for (int i = tid; i < D_; i += 256) s += p[i];
  red[tid] = s;
  __syncthreads();
  for (int st = 128; st > 0; st >>= 1) {
    if (tid < st) red[tid] += red[tid + st];
    __syncthreads();
  }
  const float mu = red[0] * (1.0f / D_);
  __syncthreads();

  float v = 0.f;
  for (int i = tid; i < D_; i += 256) {
    const float d = p[i] - mu;
    v += d * d;
  }
  red[tid] = v;
  __syncthreads();
  for (int st = 128; st > 0; st >>= 1) {
    if (tid < st) red[tid] += red[tid + st];
    __syncthreads();
  }
  const float rstd = rsqrtf(red[0] * (1.0f / D_) + EPS_);
  for (int i = tid; i < D_; i += 256)
    o[i] = (p[i] - mu) * rstd * g[i] + bb[i];
}

// ---------------------------------------------------------------------------
// Host launch
// ---------------------------------------------------------------------------
extern "C" void kernel_launch(void* const* d_in, const int* in_sizes, int n_in,
                              void* d_out, int out_size, void* d_ws,
                              size_t ws_size, hipStream_t stream) {
  (void)in_sizes; (void)n_in; (void)out_size; (void)ws_size;

  const float* Q   = (const float*)d_in[0];
  const float* K   = (const float*)d_in[1];
  const float* V   = (const float*)d_in[2];
  const unsigned char* pad = (const unsigned char*)d_in[3];
  const float* Wq  = (const float*)d_in[4];
  const float* bq  = (const float*)d_in[5];
  const float* Wk  = (const float*)d_in[6];
  const float* bk  = (const float*)d_in[7];
  const float* Wv  = (const float*)d_in[8];
  const float* bv  = (const float*)d_in[9];
  const float* Wo  = (const float*)d_in[10];
  const float* bo  = (const float*)d_in[11];
  const float* lng = (const float*)d_in[12];
  const float* lnb = (const float*)d_in[13];

  float* out  = (float*)d_out;                           // [B,S,D]
  float* attn = out + (size_t)B_ * S_ * D_;              // [B,H,S,S]

  // Workspace carve-up
  size_t off = 0;
  char* wsb = (char*)d_ws;
  auto take = [&](size_t bytes) -> void* {
    void* p = wsb + off;
    off += (bytes + 255) & ~(size_t)255;
    return p;
  };
  const size_t MD = (size_t)B_ * S_ * D_;        // 8192*768
  const size_t WW = (size_t)D_ * D_;             // 768*768
  const size_t HD = (size_t)B_ * H_ * S_ * DK_;  // head tensors

  __bf16* Xq  = (__bf16*)take(MD * 2);
  __bf16* Xk  = (__bf16*)take(MD * 2);
  __bf16* Xv  = (__bf16*)take(MD * 2);
  __bf16* Wqb = (__bf16*)take(WW * 2);
  __bf16* Wkb = (__bf16*)take(WW * 2);
  __bf16* Wvb = (__bf16*)take(WW * 2);
  __bf16* Wob = (__bf16*)take(WW * 2);
  __bf16* qh  = (__bf16*)take(HD * 2);
  __bf16* kh  = (__bf16*)take(HD * 2);
  __bf16* vh  = (__bf16*)take(HD * 2);
  __bf16* ctx = (__bf16*)take(MD * 2);
  float*  otm = (float*)take(MD * 4);

  auto cvt = [&](const float* s, __bf16* d, size_t n) {
    cvt_f32_to_bf16<<<dim3((unsigned)((n + 255) / 256)), dim3(256), 0,
                      stream>>>(s, d, (long)n);
  };
  cvt(Q, Xq, MD);
  cvt(K, Xk, MD);
  cvt(V, Xv, MD);
  cvt(Wq, Wqb, WW);
  cvt(Wk, Wkb, WW);
  cvt(Wv, Wvb, WW);
  cvt(Wo, Wob, WW);

  // QKV projections (M=8192, N=768, K=768)
  gemm_proj_head<<<dim3(D_ / 64, (B_ * S_) / 128), dim3(256), 0, stream>>>(
      Xq, Wqb, bq, qh);
  gemm_proj_head<<<dim3(D_ / 64, (B_ * S_) / 128), dim3(256), 0, stream>>>(
      Xk, Wkb, bk, kh);
  gemm_proj_head<<<dim3(D_ / 64, (B_ * S_) / 128), dim3(256), 0, stream>>>(
      Xv, Wvb, bv, vh);

  // Scores + mask (writes pre-softmax f32 into attn output region)
  attn_scores<<<dim3(S_ / 64, S_ / 128, B_ * H_), dim3(256), 0, stream>>>(
      qh, kh, pad, attn);

  // Softmax in place -> final attn output
  softmax_rows<<<dim3(B_ * H_ * S_), dim3(256), 0, stream>>>(attn);

  // Context GEMM (per (b,h): 2048x2048 @ 2048x64)
  attn_context<<<dim3(1, S_ / 128, B_ * H_), dim3(256), 0, stream>>>(attn, vh,
                                                                     ctx);

  // Output projection + bias + residual
  gemm_oproj<<<dim3(D_ / 64, (B_ * S_) / 128), dim3(256), 0, stream>>>(
      ctx, Wob, bo, Q, otm);

  // LayerNorm -> final out
  layernorm_rows<<<dim3(B_ * S_), dim3(256), 0, stream>>>(otm, lng, lnb, out);
}
